// RotaryMultiheadAttention_21509196219145
// MI455X (gfx1250) — matrix operand order repaired and checked by reading, not compiled, not run on hardware
//
#include <hip/hip_runtime.h>
#include <hip/hip_bf16.h>

// ---------------------------------------------------------------------------
// Rotary multi-head attention forward for gfx1250 (MI455X).
// bf16 WMMA (v_wmma_f32_16x16x32_bf16) for all four matmuls,
// fused RoPE in the QKV-GEMM epilogue, flash-attention with online softmax.
// Copy-free ping-pong register double buffering in all K loops.
// ---------------------------------------------------------------------------

typedef __attribute__((ext_vector_type(16))) __bf16 bf16x16;
typedef __attribute__((ext_vector_type(8)))  __bf16 bf16x8;
typedef __attribute__((ext_vector_type(8)))  float  f32x8;

#define BB 4
#define LL 2048
#define EE 1024
#define HH 16
#define HD 64
#define SCALE 0.125f   // 1/sqrt(64)

static __device__ __forceinline__ f32x8 wmma_bf16(bf16x16 a, bf16x16 b, f32x8 c) {
  return __builtin_amdgcn_wmma_f32_16x16x32_bf16(false, a, false, b, (short)0, c,
                                                 false, false);
}

// Build a 16-element bf16 fragment from two 8-element (16B) chunks.
static __device__ __forceinline__ bf16x16 ld_frag(const __bf16* p0, const __bf16* p1) {
  union { bf16x16 v; bf16x8 h[2]; } u;
  u.h[0] = *(const bf16x8*)p0;
  u.h[1] = *(const bf16x8*)p1;
  return u.v;
}

static __device__ __forceinline__ float redmax16(float v) {
  #pragma unroll
  for (int m = 1; m <= 8; m <<= 1) v = fmaxf(v, __shfl_xor(v, m, 32));
  return v;
}
static __device__ __forceinline__ float redsum16(float v) {
  #pragma unroll
  for (int m = 1; m <= 8; m <<= 1) v += __shfl_xor(v, m, 32);
  return v;
}

// Fragment set for a 32(M) x 64(N) wave tile.
struct Frags {
  bf16x16 a0, a1;
  bf16x16 b[4];
};

static __device__ __forceinline__ void load_frags(Frags& f,
    const __bf16* __restrict__ arow0, const __bf16* __restrict__ arow1,
    const __bf16* const* bcol, int k, int koffA, int koffB) {
  f.a0 = ld_frag(arow0 + k + koffA, arow0 + k + 16 + koffA);
  f.a1 = ld_frag(arow1 + k + koffA, arow1 + k + 16 + koffA);
  #pragma unroll
  for (int t = 0; t < 4; t++)
    f.b[t] = ld_frag(bcol[t] + k + koffB, bcol[t] + k + koffB + 16);
}

static __device__ __forceinline__ void frags_wmma(f32x8 acc[2][4], const Frags& f) {
  #pragma unroll
  for (int t = 0; t < 4; t++) acc[0][t] = wmma_bf16(f.a0, f.b[t], acc[0][t]);
  #pragma unroll
  for (int t = 0; t < 4; t++) acc[1][t] = wmma_bf16(f.a1, f.b[t], acc[1][t]);
}

// ---------------------------------------------------------------------------
__global__ __launch_bounds__(256) void cvt_f32_bf16(const float* __restrict__ src,
                                                    __bf16* __restrict__ dst, int n) {
  int i = blockIdx.x * blockDim.x + threadIdx.x;
  if (i < n) dst[i] = (__bf16)src[i];
}

// ---------------------------------------------------------------------------
// QKV GEMM: qkv[m,n] = sum_k x[m,k] * Wqkv[n,k] + bqkv[n]; fused RoPE + layout.
// One wave computes a 32(M) x 64(N) tile (2 M sub-tiles share B fragments).
// Q,K -> [B,H,L,64] bf16 ; V -> transposed [B,H,64,L] bf16.
__global__ __launch_bounds__(256) void qkv_rope_kernel(
    const __bf16* __restrict__ Xb, const __bf16* __restrict__ Wb,
    const float* __restrict__ bias, const float* __restrict__ cosp,
    const float* __restrict__ sinp,
    __bf16* __restrict__ Qb, __bf16* __restrict__ Kb, __bf16* __restrict__ Vtb) {
  const int lane = threadIdx.x & 31;
  const int w    = threadIdx.x >> 5;
  const int tile = blockIdx.x * 8 + w;
  const int tn = tile % 48;       // 48 N-slabs of 64 over 3E=3072
  const int tm = tile / 48;       // 256 M tiles of 32 over B*L=8192
  const int m0 = tm * 32;
  const int ng = tn * 64;
  const int nn = lane & 15;
  const int hi = lane >> 4;
  const int koffA = hi ? 8 : 0;
  const int koffB = hi ? 8 : 0;

  f32x8 acc[2][4] = {};

  const __bf16* arow0 = Xb + (size_t)(m0 + nn) * EE;
  const __bf16* arow1 = Xb + (size_t)(m0 + 16 + nn) * EE;
  const __bf16* bcol[4];
  #pragma unroll
  for (int t = 0; t < 4; t++) bcol[t] = Wb + (size_t)(ng + t * 16 + nn) * EE;

  // copy-free ping-pong: f0/f1 alternately loaded and consumed
  Frags f0, f1;
  load_frags(f0, arow0, arow1, bcol, 0, koffA, koffB);
  for (int k = 0; k < EE; k += 64) {
    __builtin_prefetch(arow0 + k + 256, 0, 1);
    load_frags(f1, arow0, arow1, bcol, k + 32, koffA, koffB);
    frags_wmma(acc, f0);
    const int k2 = (k + 64 < EE) ? (k + 64) : 0;   // clamp: last prefetch unused
    load_frags(f0, arow0, arow1, bcol, k2, koffA, koffB);
    frags_wmma(acc, f1);
  }

  // bias
  #pragma unroll
  for (int t = 0; t < 4; t++) {
    float bv = bias[ng + t * 16 + nn];
    #pragma unroll
    for (int half = 0; half < 2; half++)
      #pragma unroll
      for (int r = 0; r < 8; r++) acc[half][t][r] += bv;
  }

  const int sec  = ng >> 10;        // 0=q, 1=k, 2=v
  const int nloc = ng & 1023;
  const int h    = nloc >> 6;

  if (sec < 2) {  // RoPE: pairs (d, d+32) = (acc[.][t], acc[.][t+2]), t in {0,1}
    #pragma unroll
    for (int half = 0; half < 2; half++)
      #pragma unroll
      for (int r = 0; r < 8; r++) {
        int m = m0 + half * 16 + r + 8 * hi;
        int l = m & (LL - 1);
        #pragma unroll
        for (int t = 0; t < 2; t++) {
          int dd = t * 16 + nn;              // freq index 0..31
          float c = cosp[l * 32 + dd];
          float s = sinp[l * 32 + dd];
          float x1 = acc[half][t][r], x2 = acc[half][t + 2][r];
          acc[half][t][r]     = x1 * c - x2 * s;
          acc[half][t + 2][r] = x1 * s + x2 * c;
        }
      }
  }

  #pragma unroll
  for (int half = 0; half < 2; half++)
    #pragma unroll
    for (int r = 0; r < 8; r++) {
      int m = m0 + half * 16 + r + 8 * hi;
      int b = m >> 11;                // /L
      int l = m & (LL - 1);
      #pragma unroll
      for (int t = 0; t < 4; t++) {
        int d = t * 16 + nn;
        __bf16 v = (__bf16)acc[half][t][r];
        if (sec == 0)      Qb[((size_t)(b * HH + h) * LL + l) * HD + d] = v;
        else if (sec == 1) Kb[((size_t)(b * HH + h) * LL + l) * HD + d] = v;
        else               Vtb[((size_t)(b * HH + h) * HD + d) * LL + l] = v;
      }
    }
}

// ---------------------------------------------------------------------------
// Flash attention: one wave = 16 query rows; key loop in chunks of 32,
// unrolled x2 with copy-free ping-pong K fragments.
// S = Q K^T via 2x2 chained WMMA; online softmax; O += P V via 4 WMMA.
__global__ __launch_bounds__(256) void flash_kernel(
    const __bf16* __restrict__ Qb, const __bf16* __restrict__ Kb,
    const __bf16* __restrict__ Vtb, __bf16* __restrict__ AO) {
  __shared__ alignas(16) __bf16 Pbuf[8 * 16 * 32];

  const int lane = threadIdx.x & 31;
  const int w    = threadIdx.x >> 5;
  const int bh   = blockIdx.x >> 4;    // B*H = 64
  const int qblk = blockIdx.x & 15;    // 16 q-blocks of 128
  const int b = bh >> 4, h = bh & 15;
  const int q0 = qblk * 128 + w * 16;
  const int nn = lane & 15;
  const int hi = lane >> 4;
  const int koffA = hi ? 8 : 0;
  const int koffB = hi ? 8 : 0;

  const __bf16* Qbase = Qb + (size_t)bh * LL * HD;
  const __bf16* Kbase = Kb + (size_t)bh * LL * HD;
  const __bf16* Vbase = Vtb + (size_t)bh * HD * LL;

  // Q fragments (constant over key loop): a0 = d0..31, a1 = d32..63
  const __bf16* qrow = Qbase + (size_t)(q0 + nn) * HD;
  bf16x16 a0 = ld_frag(qrow + koffA,      qrow + 16 + koffA);
  bf16x16 a1 = ld_frag(qrow + 32 + koffA, qrow + 48 + koffA);

  float mrow[8], lrow[8];
  f32x8 o[4] = {};
  #pragma unroll
  for (int r = 0; r < 8; r++) { mrow[r] = -1e30f; lrow[r] = 0.f; }

  __bf16* pb = Pbuf + w * 512;   // wave-private 16x32 bf16 tile

  auto load_k = [&](bf16x16 (&kf)[2][2], int kt) {
    #pragma unroll
    for (int c = 0; c < 2; c++) {
      const __bf16* kp = Kbase + (size_t)(kt + c * 16 + nn) * HD + koffB;
      kf[c][0] = ld_frag(kp,      kp + 16);
      kf[c][1] = ld_frag(kp + 32, kp + 48);
    }
  };

  // one 32-key chunk: consume kcur, then prefetch knext@ktn during softmax
  auto chunk = [&](int kt, bf16x16 (&kcur)[2][2], bf16x16 (&knext)[2][2], int ktn) {
    // V fragments for this chunk, issued early
    bf16x16 vf[4];
    #pragma unroll
    for (int t = 0; t < 4; t++) {
      const __bf16* vp = Vbase + (size_t)(t * 16 + nn) * LL + kt + koffB;
      vf[t] = ld_frag(vp, vp + 16);
    }
    // S tiles: two 16-key sub-tiles
    f32x8 s[2];
    #pragma unroll
    for (int c = 0; c < 2; c++) {
      f32x8 z = {};
      s[c] = wmma_bf16(a0, kcur[c][0], z);
      s[c] = wmma_bf16(a1, kcur[c][1], s[c]);
      s[c] *= SCALE;
    }
    // next chunk's K fragments (latency hides under softmax VALU)
    load_k(knext, ktn);
    // online softmax
    #pragma unroll
    for (int r = 0; r < 8; r++) {
      float rm = redmax16(fmaxf(s[0][r], s[1][r]));
      float mn = fmaxf(mrow[r], rm);
      float alpha = __expf(mrow[r] - mn);
      mrow[r] = mn;
      float p0 = __expf(s[0][r] - mn);
      float p1 = __expf(s[1][r] - mn);
      lrow[r] = lrow[r] * alpha + redsum16(p0 + p1);
      s[0][r] = p0; s[1][r] = p1;
      #pragma unroll
      for (int t = 0; t < 4; t++) o[t][r] *= alpha;
    }
    // P relayout through LDS into A-fragment striping
    #pragma unroll
    for (int c = 0; c < 2; c++)
      #pragma unroll
      for (int r = 0; r < 8; r++)
        pb[(r + 8 * hi) * 32 + c * 16 + nn] = (__bf16)s[c][r];
    const __bf16* pr = pb + nn * 32 + (hi ? 8 : 0);
    bf16x16 pa = ld_frag(pr, pr + 16);
    // O += P * V (V stored d-major -> contiguous B-fragments)
    #pragma unroll
    for (int t = 0; t < 4; t++) o[t] = wmma_bf16(pa, vf[t], o[t]);
  };

  bf16x16 kfA[2][2], kfB[2][2];
  load_k(kfA, 0);
  for (int kt = 0; kt < LL; kt += 64) {
    chunk(kt,      kfA, kfB, kt + 32);
    chunk(kt + 32, kfB, kfA, (kt + 64 < LL) ? (kt + 64) : 0);
  }

  // ---- normalize + store to [B*L, E] bf16 ----
  #pragma unroll
  for (int r = 0; r < 8; r++) {
    float inv = 1.f / lrow[r];
    int grow = b * LL + q0 + r + 8 * hi;
    #pragma unroll
    for (int t = 0; t < 4; t++) {
      int col = h * HD + t * 16 + nn;
      AO[(size_t)grow * EE + col] = (__bf16)(o[t][r] * inv);
    }
  }
}

// ---------------------------------------------------------------------------
// Output projection: out[m,n] = sum_k AO[m,k] * Wout[n,k] + bout[n]  (f32 out)
// 32(M) x 64(N) per wave, copy-free ping-pong K loop.
__global__ __launch_bounds__(256) void out_proj_kernel(
    const __bf16* __restrict__ Ab, const __bf16* __restrict__ Wb,
    const float* __restrict__ bias, float* __restrict__ out) {
  const int lane = threadIdx.x & 31;
  const int w    = threadIdx.x >> 5;
  const int tile = blockIdx.x * 8 + w;
  const int tn = tile & 15;       // 16 N-slabs of 64 over E=1024
  const int tm = tile >> 4;       // 256 M tiles of 32
  const int m0 = tm * 32;
  const int n0 = tn * 64;
  const int nn = lane & 15;
  const int hi = lane >> 4;
  const int koffA = hi ? 8 : 0;
  const int koffB = hi ? 8 : 0;

  f32x8 acc[2][4] = {};
  const __bf16* arow0 = Ab + (size_t)(m0 + nn) * EE;
  const __bf16* arow1 = Ab + (size_t)(m0 + 16 + nn) * EE;
  const __bf16* bcol[4];
  #pragma unroll
  for (int t = 0; t < 4; t++) bcol[t] = Wb + (size_t)(n0 + t * 16 + nn) * EE;

  Frags f0, f1;
  load_frags(f0, arow0, arow1, bcol, 0, koffA, koffB);
  for (int k = 0; k < EE; k += 64) {
    __builtin_prefetch(arow0 + k + 256, 0, 1);
    load_frags(f1, arow0, arow1, bcol, k + 32, koffA, koffB);
    frags_wmma(acc, f0);
    const int k2 = (k + 64 < EE) ? (k + 64) : 0;
    load_frags(f0, arow0, arow1, bcol, k2, koffA, koffB);
    frags_wmma(acc, f1);
  }

  #pragma unroll
  for (int t = 0; t < 4; t++) {
    float bv = bias[n0 + t * 16 + nn];
    #pragma unroll
    for (int half = 0; half < 2; half++)
      #pragma unroll
      for (int r = 0; r < 8; r++) {
        int m = m0 + half * 16 + r + 8 * hi;
        out[(size_t)m * EE + n0 + t * 16 + nn] = acc[half][t][r] + bv;
      }
  }
}

// ---------------------------------------------------------------------------
extern "C" void kernel_launch(void* const* d_in, const int* in_sizes, int n_in,
                              void* d_out, int out_size, void* d_ws, size_t ws_size,
                              hipStream_t stream) {
  const float* x    = (const float*)d_in[0];
  const float* cosp = (const float*)d_in[1];
  const float* sinp = (const float*)d_in[2];
  const float* Wqkv = (const float*)d_in[3];
  const float* bqkv = (const float*)d_in[4];
  const float* Wout = (const float*)d_in[5];
  const float* bout = (const float*)d_in[6];
  float* out = (float*)d_out;

  const int NX  = BB * LL * EE;      // 8388608
  const int NWQ = 3 * EE * EE;       // 3145728
  const int NWO = EE * EE;           // 1048576
  const size_t NQ = (size_t)BB * HH * LL * HD;  // 8388608

  uint8_t* ws = (uint8_t*)d_ws;
  __bf16* Xb   = (__bf16*)(ws);
  __bf16* Wqb  = (__bf16*)(ws + (size_t)NX * 2);
  __bf16* Wob  = (__bf16*)(ws + (size_t)NX * 2 + (size_t)NWQ * 2);
  __bf16* Qb   = (__bf16*)(ws + (size_t)NX * 2 + (size_t)NWQ * 2 + (size_t)NWO * 2);
  __bf16* Kb   = Qb + NQ;
  __bf16* Vtb  = Kb + NQ;
  __bf16* AO   = Vtb + NQ;

  // 1) convert inputs to bf16
  cvt_f32_bf16<<<(NX  + 255) / 256, 256, 0, stream>>>(x,    Xb,  NX);
  cvt_f32_bf16<<<(NWQ + 255) / 256, 256, 0, stream>>>(Wqkv, Wqb, NWQ);
  cvt_f32_bf16<<<(NWO + 255) / 256, 256, 0, stream>>>(Wout, Wob, NWO);

  // 2) QKV GEMM + RoPE + layout: 256 M-tiles * 48 N-slabs / 8 waves
  qkv_rope_kernel<<<1536, 256, 0, stream>>>(Xb, Wqb, bqkv, cosp, sinp, Qb, Kb, Vtb);

  // 3) flash attention: (B*H=64) * (L/128=16) blocks
  flash_kernel<<<1024, 256, 0, stream>>>(Qb, Kb, Vtb, AO);

  // 4) output projection: 256 * 16 tiles / 8 waves
  out_proj_kernel<<<512, 256, 0, stream>>>(AO, Wob, bout, out);
}
